// SCINet_67680094650921
// MI455X (gfx1250) — matrix operand order repaired and checked by reading, not compiled, not run on hardware
//
#include <hip/hip_runtime.h>
#include <cstdint>
#include <cstddef>

// ---------------------------------------------------------------------------
// SCINet on gfx1250: all convolutions + projection as bf16 WMMA GEMMs
// (V_WMMA_F32_16X16X32_BF16, f32 accumulate), im2col-free layouts.
// ---------------------------------------------------------------------------

typedef __attribute__((ext_vector_type(16))) __bf16 v16bf;
typedef __attribute__((ext_vector_type(8)))  __bf16 v8bf;
typedef __attribute__((ext_vector_type(8)))  float  v8f;

__device__ __forceinline__ v16bf cat8(v8bf lo, v8bf hi) {
  return __builtin_shufflevector(lo, hi, 0,1,2,3,4,5,6,7,8,9,10,11,12,13,14,15);
}

// ---------------- weight prep kernels ----------------
// W1 (nb,768,192,5) [co][ci][k] -> W1R (nb,768,960) [co][k*192+ci], bf16
__global__ void reorder_w1_kernel(const float* __restrict__ src, __bf16* __restrict__ dst, long total) {
  long idx = (long)blockIdx.x * blockDim.x + threadIdx.x;
  if (idx >= total) return;
  int kc = (int)(idx % 960);
  long r = idx / 960;
  int co = (int)(r % 768);
  long nb = r / 768;
  int k = kc / 192, ci = kc % 192;
  dst[idx] = (__bf16)src[((nb * 768 + co) * 192 + ci) * 5 + k];
}

// W2 (nb,192,768,3) [d][ci][k] -> W2R (nb,192,2304) [d][k*768+ci], bf16
__global__ void reorder_w2_kernel(const float* __restrict__ src, __bf16* __restrict__ dst, long total) {
  long idx = (long)blockIdx.x * blockDim.x + threadIdx.x;
  if (idx >= total) return;
  int kc = (int)(idx % 2304);
  long r = idx / 2304;
  int d = (int)(r % 192);
  long nb = r / 192;
  int k = kc / 768, ci = kc % 768;
  dst[idx] = (__bf16)src[((nb * 192 + d) * 768 + ci) * 3 + k];
}

__global__ void cvt_bf16_kernel(const float* __restrict__ src, __bf16* __restrict__ dst, long total) {
  long idx = (long)blockIdx.x * blockDim.x + threadIdx.x;
  if (idx >= total) return;
  dst[idx] = (__bf16)src[idx];
}

// ---------------- main GEMM / conv kernel ----------------
// D[m][n] = act( sum_k A[m][k] * X[n*ldb + k] + bias[m] )
//   conv1: A=W1R, X=padded input rows, m=co(768), n=time, Kc=960,  ldb=192
//   conv2: A=W2R, X=hidden rows,       m=d (192), n=time, Kc=2304, ldb=768
//   proj : A=Wproj, X=sT rows,         m=o (512), n=d,    Kc=1024, ldb=1024
// Wave computes 32(m) x 64(n); block = WAVES waves stacked along m.
// ACT: 0=leaky(0.01), 1=tanh, 2=none.  OUTMODE: 0 -> bf16 Y[n*ldy+m], 1 -> f32 Y[m*ldy+n]
template<int WAVES, int ACT, int OUTMODE>
__global__ __launch_bounds__(WAVES * 32)
void gemm_bf16_kernel(const __bf16* __restrict__ A, const float* __restrict__ bias,
                      const __bf16* __restrict__ X, void* __restrict__ Yv,
                      int M, int Kc, int Nout, int ldb, int ldy,
                      long xz, long yz, long abr, int bbr, int numB)
{
  const int lane = threadIdx.x & 31;
  const int wave = threadIdx.x >> 5;
  const int half = lane >> 4;
  const int l15  = lane & 15;
  const int mbase = (blockIdx.y * WAVES + wave) * 32;
  const int nbase = blockIdx.x * 64;
  const int z  = blockIdx.z;
  const int br = z / numB;

  const __bf16* Ab = A + (long)br * abr;
  const __bf16* Xb = X + (long)z * xz;

  v8f acc[2][4];
  const v8f vzero = {0.f,0.f,0.f,0.f,0.f,0.f,0.f,0.f};
  #pragma unroll
  for (int mt = 0; mt < 2; ++mt)
    #pragma unroll
    for (int nt = 0; nt < 4; ++nt) acc[mt][nt] = vzero;

  // A fragment base: lane row = mbase+mt*16+(lane&15); chunks at kb+8*half and +16
  const __bf16* ap[2];
  #pragma unroll
  for (int mt = 0; mt < 2; ++mt) {
    int row = mbase + mt * 16 + l15; row = row < (M - 1) ? row : (M - 1);
    ap[mt] = Ab + (long)row * Kc + half * 8;
  }
  // B fragment base: lane col = nbase+nt*16+(lane&15); chunks at kb+8*half and +16
  const __bf16* bp[4];
  #pragma unroll
  for (int nt = 0; nt < 4; ++nt) {
    int col = nbase + nt * 16 + l15; col = col < (Nout - 1) ? col : (Nout - 1);
    bp[nt] = Xb + (long)col * ldb + half * 8;
  }

  for (int kb = 0; kb < Kc; kb += 32) {
    v16bf Am[2], Bm[4];
    #pragma unroll
    for (int mt = 0; mt < 2; ++mt) {
      v8bf lo = *(const v8bf*)(ap[mt] + kb);
      v8bf hi = *(const v8bf*)(ap[mt] + kb + 16);
      Am[mt] = cat8(lo, hi);
    }
    #pragma unroll
    for (int nt = 0; nt < 4; ++nt) {
      v8bf lo = *(const v8bf*)(bp[nt] + kb);
      v8bf hi = *(const v8bf*)(bp[nt] + kb + 16);
      Bm[nt] = cat8(lo, hi);
    }
    #pragma unroll
    for (int mt = 0; mt < 2; ++mt)
      #pragma unroll
      for (int nt = 0; nt < 4; ++nt)
        acc[mt][nt] = __builtin_amdgcn_wmma_f32_16x16x32_bf16(
            false, Am[mt], false, Bm[nt], (short)0, acc[mt][nt], false, false);
  }

  // Epilogue: C/D layout: VGPR r -> rows {m0+r}, m0 = tile base + 8*half, col = lane&15
  #pragma unroll
  for (int mt = 0; mt < 2; ++mt) {
    const int m0 = mbase + mt * 16 + half * 8;
    float bv[8];
    if (bias) {
      const float4 blo = *(const float4*)(bias + (long)br * bbr + m0);
      const float4 bhi = *(const float4*)(bias + (long)br * bbr + m0 + 4);
      bv[0]=blo.x; bv[1]=blo.y; bv[2]=blo.z; bv[3]=blo.w;
      bv[4]=bhi.x; bv[5]=bhi.y; bv[6]=bhi.z; bv[7]=bhi.w;
    } else {
      #pragma unroll
      for (int r = 0; r < 8; ++r) bv[r] = 0.f;
    }
    #pragma unroll
    for (int nt = 0; nt < 4; ++nt) {
      const int n = nbase + nt * 16 + l15;
      float vals[8];
      #pragma unroll
      for (int r = 0; r < 8; ++r) {
        float v = acc[mt][nt][r] + bv[r];
        if (ACT == 0)      v = v > 0.f ? v : 0.01f * v;
        else if (ACT == 1) v = tanhf(v);
        vals[r] = v;
      }
      if (n < Nout) {
        if (OUTMODE == 0) {
          v8bf pk;
          #pragma unroll
          for (int r = 0; r < 8; ++r) pk[r] = (__bf16)vals[r];
          __bf16* Y = (__bf16*)Yv + (long)z * yz;
          *(v8bf*)(Y + (long)n * ldy + m0) = pk;
        } else {
          float* Y = (float*)Yv + (long)z * yz;
          #pragma unroll
          for (int r = 0; r < 8; ++r) Y[(long)(m0 + r) * ldy + n] = vals[r];
        }
      }
    }
  }
}

// ---------------- elementwise kernels ----------------
// X (B, 2L, D) f32 -> XP [2][B][L+6][D] bf16 with even/odd split + edge pad(3)
__global__ void split_pad_kernel(const float* __restrict__ X, __bf16* __restrict__ XP,
                                 int Bn, int L, int Dn) {
  long idx = (long)blockIdx.x * blockDim.x + threadIdx.x;
  long total = 2L * Bn * (L + 6) * Dn;
  if (idx >= total) return;
  int d = (int)(idx % Dn);
  long r = idx / Dn;
  int t = (int)(r % (L + 6)); r /= (L + 6);
  int b = (int)(r % Bn);
  int br = (int)(r / Bn);
  int tsrc = t - 3; tsrc = tsrc < 0 ? 0 : (tsrc > L - 1 ? L - 1 : tsrc);
  float v = X[((long)b * (2 * L) + 2 * tsrc + br) * Dn + d];
  XP[idx] = (__bf16)v;
}

// d = xo*exp(phi(xe)); c = xe*exp(psi(xo)); store f32 seqs + padded bf16 for phase B
__global__ void combine1_kernel(const float* __restrict__ X, const __bf16* __restrict__ Y,
                                float* __restrict__ dseq, float* __restrict__ cseq,
                                __bf16* __restrict__ XP, int Bn, int L, int Dn) {
  long idx = (long)blockIdx.x * blockDim.x + threadIdx.x;
  long total = (long)Bn * (L + 6) * Dn;
  if (idx >= total) return;
  int d = (int)(idx % Dn);
  long r = idx / Dn;
  int t = (int)(r % (L + 6));
  int b = (int)(r / (L + 6));
  int tsrc = t - 3; tsrc = tsrc < 0 ? 0 : (tsrc > L - 1 ? L - 1 : tsrc);
  long yoff = ((long)b * L + tsrc) * Dn + d;
  float yphi = (float)Y[yoff];
  float ypsi = (float)Y[(long)Bn * L * Dn + yoff];
  float xe = X[((long)b * 2 * L + 2 * tsrc) * Dn + d];
  float xo = X[((long)b * 2 * L + 2 * tsrc + 1) * Dn + d];
  float dv = xo * expf(yphi);
  float cv = xe * expf(ypsi);
  long po = ((long)b * (L + 6) + t) * Dn + d;
  XP[po] = (__bf16)dv;
  XP[(long)Bn * (L + 6) * Dn + po] = (__bf16)cv;
  if (t >= 3 && t < L + 3) {
    long so = ((long)b * L + (t - 3)) * Dn + d;
    dseq[so] = dv;
    cseq[so] = cv;
  }
}

// even = c + eta(d); odd = d - rho(c); write with (BT, offset, stride) -> folds zip-up
__global__ void combine2_kernel(const __bf16* __restrict__ Y, const float* __restrict__ dseq,
                                const float* __restrict__ cseq, float* __restrict__ evB,
                                float* __restrict__ odB, int Bn, int L, int Dn,
                                int BT, int evOff, int odOff, int strd) {
  long idx = (long)blockIdx.x * blockDim.x + threadIdx.x;
  long total = (long)Bn * L * Dn;
  if (idx >= total) return;
  int d = (int)(idx % Dn);
  long r = idx / Dn;
  int t = (int)(r % L);
  int b = (int)(r / L);
  long so = ((long)b * L + t) * Dn + d;
  float yeta = (float)Y[so];
  float yrho = (float)Y[(long)Bn * L * Dn + so];
  float ev = cseq[so] + yeta;
  float od = dseq[so] - yrho;
  evB[((long)b * BT + evOff + (long)t * strd) * Dn + d] = ev;
  odB[((long)b * BT + odOff + (long)t * strd) * Dn + d] = od;
}

// sT[b][d][t] = bf16(tree[b][t][d] + x[b][t][d])  (transpose so proj B-cols are contiguous)
__global__ void resid_transpose_kernel(const float* __restrict__ tree, const float* __restrict__ x,
                                       __bf16* __restrict__ sT, int Bn, int Tn, int Dn) {
  long idx = (long)blockIdx.x * blockDim.x + threadIdx.x;
  long total = (long)Bn * Tn * Dn;
  if (idx >= total) return;
  int d = (int)(idx % Dn);
  long r = idx / Dn;
  int t = (int)(r % Tn);
  int b = (int)(r / Tn);
  float v = tree[idx] + x[idx];
  sT[((long)b * Dn + d) * Tn + t] = (__bf16)v;
}

// ---------------------------------------------------------------------------
extern "C" void kernel_launch(void* const* d_in, const int* in_sizes, int n_in,
                              void* d_out, int out_size, void* d_ws, size_t ws_size,
                              hipStream_t stream) {
  (void)in_sizes; (void)n_in; (void)out_size; (void)ws_size;
  const float* x   = (const float*)d_in[0];
  const float* W1  = (const float*)d_in[1];
  const float* b1  = (const float*)d_in[2];
  const float* W2  = (const float*)d_in[3];
  const float* b2  = (const float*)d_in[4];
  const float* Wpj = (const float*)d_in[5];

  const int Bn = 32, Tn = 1024, Dn = 192, OUTL = 512;

  char* ws = (char*)d_ws;
  size_t off = 0;
  auto carve = [&](size_t bytes) -> void* {
    void* p = ws + off;
    off += (bytes + 255) & ~(size_t)255;
    return p;
  };

  __bf16* W1R  = (__bf16*)carve((size_t)28 * 768 * 960 * 2);
  __bf16* W2R  = (__bf16*)carve((size_t)28 * 192 * 2304 * 2);
  __bf16* WPR  = (__bf16*)carve((size_t)OUTL * Tn * 2);
  __bf16* XPAD = (__bf16*)carve((size_t)2 * Bn * 518 * Dn * 2);
  __bf16* HB   = (__bf16*)carve((size_t)2 * Bn * 514 * 768 * 2);
  __bf16* YBR  = (__bf16*)carve((size_t)2 * Bn * 512 * Dn * 2);
  float* DSEQ  = (float*)carve((size_t)Bn * 512 * Dn * 4);
  float* CSEQ  = (float*)carve((size_t)Bn * 512 * Dn * 4);
  float* Ebuf  = (float*)carve((size_t)Bn * 512 * Dn * 4);
  float* Obuf  = (float*)carve((size_t)Bn * 512 * Dn * 4);
  float* EE    = (float*)carve((size_t)Bn * 256 * Dn * 4);
  float* EO    = (float*)carve((size_t)Bn * 256 * Dn * 4);
  float* OE    = (float*)carve((size_t)Bn * 256 * Dn * 4);
  float* OO    = (float*)carve((size_t)Bn * 256 * Dn * 4);
  float* TREE  = (float*)carve((size_t)Bn * Tn * Dn * 4);
  __bf16* ST   = (__bf16*)carve((size_t)Bn * Dn * Tn * 2);

  auto blocks = [](long total) { return (unsigned)((total + 255) / 256); };

  { long t = 28L * 768 * 960;  reorder_w1_kernel<<<blocks(t), 256, 0, stream>>>(W1, W1R, t); }
  { long t = 28L * 192 * 2304; reorder_w2_kernel<<<blocks(t), 256, 0, stream>>>(W2, W2R, t); }
  { long t = (long)OUTL * Tn;  cvt_bf16_kernel<<<blocks(t), 256, 0, stream>>>(Wpj, WPR, t); }

  auto run_node = [&](const float* Xin, int L, int widx,
                      float* evB, float* odB, int BT, int evOff, int odOff, int strd) {
    { long t = 2L * Bn * (L + 6) * Dn;
      split_pad_kernel<<<blocks(t), 256, 0, stream>>>(Xin, XPAD, Bn, L, Dn); }
    for (int ph = 0; ph < 2; ++ph) {
      const __bf16* w1p = W1R + (size_t)(widx * 4 + ph * 2) * 768 * 960;
      const float*  b1p = b1  + (size_t)(widx * 4 + ph * 2) * 768;
      const __bf16* w2p = W2R + (size_t)(widx * 4 + ph * 2) * 192 * 2304;
      const float*  b2p = b2  + (size_t)(widx * 4 + ph * 2) * 192;
      // conv1: M=768 (3 blocks x 8 waves x 32), N=L+2, K=960, leaky, out bf16 [t][co]
      dim3 g1((unsigned)((L + 2 + 63) / 64), 3, 2 * Bn);
      gemm_bf16_kernel<8, 0, 0><<<g1, 256, 0, stream>>>(
          w1p, b1p, XPAD, HB, 768, 960, L + 2, 192, 768,
          (long)(L + 6) * 192, (long)(L + 2) * 768, 768L * 960, 768, Bn);
      // conv2: M=192 (1 block x 6 waves x 32), N=L, K=2304, tanh, out bf16 [t][d]
      dim3 g2((unsigned)((L + 63) / 64), 1, 2 * Bn);
      gemm_bf16_kernel<6, 1, 0><<<g2, 192, 0, stream>>>(
          w2p, b2p, HB, YBR, 192, 2304, L, 768, 192,
          (long)(L + 2) * 768, (long)L * 192, 192L * 2304, 192, Bn);
      if (ph == 0) {
        long t = (long)Bn * (L + 6) * Dn;
        combine1_kernel<<<blocks(t), 256, 0, stream>>>(Xin, YBR, DSEQ, CSEQ, XPAD, Bn, L, Dn);
      } else {
        long t = (long)Bn * L * Dn;
        combine2_kernel<<<blocks(t), 256, 0, stream>>>(YBR, DSEQ, CSEQ, evB, odB,
                                                       Bn, L, Dn, BT, evOff, odOff, strd);
      }
    }
  };

  // Tree (pre-order indices match reference): leaves write zip-folded into TREE.
  run_node(x,    512, 0, Ebuf, Obuf, 512, 0, 0, 1);
  run_node(Ebuf, 256, 1, EE, EO, 256, 0, 0, 1);
  run_node(EE,   128, 2, TREE, TREE, 1024, 0, 4, 8);
  run_node(EO,   128, 3, TREE, TREE, 1024, 2, 6, 8);
  run_node(Obuf, 256, 4, OE, OO, 256, 0, 0, 1);
  run_node(OE,   128, 5, TREE, TREE, 1024, 1, 5, 8);
  run_node(OO,   128, 6, TREE, TREE, 1024, 3, 7, 8);

  { long t = (long)Bn * Tn * Dn;
    resid_transpose_kernel<<<blocks(t), 256, 0, stream>>>(TREE, x, ST, Bn, Tn, Dn); }

  // projection: M=512 (2 blocks x 8 waves), N=192, K=1024, no act, f32 out [o][d]
  dim3 gp(3, 2, Bn);
  gemm_bf16_kernel<8, 2, 1><<<gp, 256, 0, stream>>>(
      WPR, nullptr, ST, d_out, 512, 1024, 192, 1024, 192,
      (long)Dn * Tn, (long)OUTL * Dn, 0L, 0, Bn);
}
